// GAT_Critic_34187939676289
// MI455X (gfx1250) — matrix operand order repaired and checked by reading, not compiled, not run on hardware
//
#include <hip/hip_runtime.h>
#include <hip/hip_bf16.h>
#include <stdint.h>

// ---------------- problem constants ----------------
#define BB   16          // graphs
#define NN   1000        // nodes per graph
#define EE   16000       // edges per graph
#define NT   (BB*NN)     // 16000 total nodes
#define BEc  (BB*EE)     // 256000 batched edges
#define ETc  (BEc+NT)    // 272000 edges incl self loops
#define F1   1024        // H1*C1
#define H1n  8
#define C1n  128
#define EMBn 64
#define HIDn 128

// ---------------- WMMA vector types ----------------
typedef __bf16 bf16_t;
typedef __attribute__((ext_vector_type(16))) bf16_t         v16bf;
typedef __attribute__((ext_vector_type(16))) unsigned short v16us;
typedef __attribute__((ext_vector_type(8)))  float          v8f;

// ---------------- TDM support ----------------
#if __has_builtin(__builtin_amdgcn_tensor_load_to_lds)
#define HAVE_TDM 1
#endif
typedef __attribute__((ext_vector_type(4))) unsigned int u32x4;
typedef __attribute__((ext_vector_type(4))) int          i32x4;
typedef __attribute__((ext_vector_type(8))) int          i32x8;

#ifdef HAVE_TDM
// Build a 2-D tensor DMA descriptor per CDNA5 ISA §8 and issue TENSOR_LOAD_TO_LDS.
// dims/strides in elements of 2 bytes (data_size code 1). Uniform args only.
__device__ __forceinline__ void tdm_load_2d(unsigned lds_off, const void* gptr,
                                            unsigned tensor_d0, unsigned tensor_d1,
                                            unsigned tile_d0, unsigned tile_d1,
                                            unsigned stride0) {
  unsigned long long ga = (unsigned long long)(uintptr_t)gptr;
  u32x4 g0;
  g0[0] = 1u;                                             // count=1, no gather mode
  g0[1] = lds_off;                                        // LDS byte address
  g0[2] = (unsigned)(ga & 0xFFFFFFFFu);                   // global_addr[31:0]
  g0[3] = (unsigned)((ga >> 32) & 0x01FFFFFFu) | (2u << 30); // addr[56:32] | type=2
  i32x8 g1;
  g1[0] = (int)(1u << 16);                                // wg_mask=0, data_size=1 (2B)
  g1[1] = (int)((tensor_d0 & 0xFFFFu) << 16);             // barrier_addr=0 | dim0[15:0]
  g1[2] = (int)(((tensor_d0 >> 16) & 0xFFFFu) | ((tensor_d1 & 0xFFFFu) << 16));
  g1[3] = (int)(((tensor_d1 >> 16) & 0xFFFFu) | (tile_d0 << 16));
  g1[4] = (int)(tile_d1 & 0xFFFFu);                       // tile_dim1 | tile_dim2=0
  g1[5] = (int)stride0;                                   // dim0_stride[31:0]
  g1[6] = 0;                                              // dim0_stride[47:32] | dim1_stride lo
  g1[7] = 0;
  i32x4 gz = {0, 0, 0, 0};
#if __clang_major__ >= 23
  i32x8 gz8 = {0, 0, 0, 0, 0, 0, 0, 0};
  __builtin_amdgcn_tensor_load_to_lds(g0, g1, gz, gz, gz8, 0);
#else
  __builtin_amdgcn_tensor_load_to_lds(g0, g1, gz, gz, 0);
#endif
}
#endif

// ---------------- workspace layout (bytes) ----------------
static constexpr size_t SZ_REG    = (size_t)NT * F1 * 4;          // 65,536,000
static constexpr size_t OFS_A     = 0;                            // h1; reused x2bf, w1bf
static constexpr size_t OFS_B     = OFS_A + SZ_REG;               // out1; reused h2/out2/x3bf
static constexpr size_t OFS_ASRC1 = OFS_B + SZ_REG;               // NT*8 f32
static constexpr size_t OFS_ADST1 = OFS_ASRC1 + 512000;
static constexpr size_t OFS_AMAX1 = OFS_ADST1 + 512000;           // NT*8 u32
static constexpr size_t OFS_DEN1  = OFS_AMAX1 + 512000;
static constexpr size_t OFS_EIS   = OFS_DEN1  + 512000;           // ET i32
static constexpr size_t OFS_EID   = OFS_EIS   + 1088000;
static constexpr size_t OFS_W2BF  = OFS_EID   + 1088000;          // 1024*64 bf16 (interleaved)
static constexpr size_t OFS_Y1ACC = OFS_W2BF  + 131072;           // 16*128 f32
static constexpr size_t OFS_Y1    = OFS_Y1ACC + 8192;
static constexpr size_t OFS_Y2    = OFS_Y1    + 8192;
static constexpr size_t OFS_X2BF  = OFS_A;                        // NT*1024 bf16
static constexpr size_t OFS_W1BF  = OFS_A + 40000000;             // 64000*128 bf16 (interleaved)
static constexpr size_t OFS_H2    = OFS_B;                        // NT*64 f32
static constexpr size_t OFS_OUT2  = OFS_B + 4194304;              // NT*64 f32
static constexpr size_t OFS_X3BF  = OFS_B + 8388608;              // NT*64 bf16

// ---------------- helpers ----------------
__device__ __forceinline__ unsigned f2ord(float f) {              // order-preserving map
  unsigned u = __float_as_uint(f);
  return (u & 0x80000000u) ? ~u : (u | 0x80000000u);
}
__device__ __forceinline__ float ord2f(unsigned u) {
  unsigned b = (u & 0x80000000u) ? (u & 0x7FFFFFFFu) : ~u;
  return __uint_as_float(b);
}
__device__ __forceinline__ unsigned short f2bf(float f) {         // RNE f32->bf16
  unsigned u = __float_as_uint(f);
  u += 0x7FFFu + ((u >> 16) & 1u);
  return (unsigned short)(u >> 16);
}
__device__ __forceinline__ float elu1(float v)  { return v > 0.f ? v : (__expf(v) - 1.f); }
__device__ __forceinline__ float lrelu(float v) { return v > 0.f ? v : 0.2f * v; }

// ---------------- generic kernels ----------------
__global__ void zero_f32(float* __restrict__ p, long n) {
  long t = (long)blockIdx.x * blockDim.x + threadIdx.x;
  if (t < n) p[t] = 0.f;
}
// f32 -> bf16 with K-pair interleave: (k,col) -> (k/2)*2*cols + col*2 + (k&1),
// so a B-matrix WMMA fragment pair (K,K+1) is one aligned 32-bit load.
__global__ void cvt_bf16_pair(const float* __restrict__ s, unsigned short* __restrict__ d,
                              long n, int colshift) {
  long t = (long)blockIdx.x * blockDim.x + threadIdx.x;
  if (t >= n) return;
  long k  = t >> colshift;
  int col = (int)(t & ((1 << colshift) - 1));
  d[((k >> 1) << (colshift + 1)) + (col << 1) + (k & 1)] = f2bf(s[t]);
}

// ---------------- conv1: h1 = x @ W1 (K=3, pure VALU) ----------------
__global__ void h1_kernel(const float* __restrict__ actions, const float* __restrict__ nodef,
                          const float* __restrict__ W1, float* __restrict__ h1) {
  long t = (long)blockIdx.x * blockDim.x + threadIdx.x;           // NT*1024
  if (t >= (long)NT * F1) return;
  int n = (int)(t >> 10);
  int j = (int)(t & 1023);
  float x0 = actions[(size_t)n * 2 + 0];
  float x1 = actions[(size_t)n * 2 + 1];
  float x2 = nodef[n];                                            // [B,1,N] flat == [B*N]
  h1[t] = fmaf(x0, W1[j], fmaf(x1, W1[F1 + j], x2 * W1[2 * F1 + j]));
}

__global__ void att1_kernel(const float* __restrict__ h1, const float* __restrict__ as,
                            const float* __restrict__ ad, float* __restrict__ osrc,
                            float* __restrict__ odst) {
  int t = blockIdx.x * blockDim.x + threadIdx.x;                  // NT*8
  if (t >= NT * H1n) return;
  int n = t >> 3, h = t & 7;
  const float* hp = h1 + (size_t)n * F1 + h * C1n;
  float s1 = 0.f, s2 = 0.f;
  #pragma unroll 8
  for (int c = 0; c < C1n; ++c) {
    float v = hp[c];
    s1 = fmaf(v, as[h * C1n + c], s1);
    s2 = fmaf(v, ad[h * C1n + c], s2);
  }
  osrc[t] = s1; odst[t] = s2;
}

// pass 1: decode faithful reshape(2,-1) edge indices, segment-max of leaky_relu scores
__global__ void edge_max1(const int* __restrict__ ei, const float* __restrict__ asrc,
                          const float* __restrict__ adst, int* __restrict__ eis,
                          int* __restrict__ eid, unsigned* __restrict__ amax) {
  long j = (long)blockIdx.x * blockDim.x + threadIdx.x;
  if (j >= ETc) return;
  int s, d;
  if (j < BEc) {
    long b  = j / (2L * EE);  long r0 = j - b * (2L * EE);
    int  r  = (int)(r0 / EE); int  e  = (int)(r0 - (long)r * EE);
    s = ei[(b * 2 + r) * EE + e] + (int)(b * NN);
    long j2 = (long)BEc + j;
    long b2 = j2 / (2L * EE); long r1 = j2 - b2 * (2L * EE);
    int  r2 = (int)(r1 / EE); int  e2 = (int)(r1 - (long)r2 * EE);
    d = ei[(b2 * 2 + r2) * EE + e2] + (int)(b2 * NN);
  } else { s = d = (int)(j - BEc); }
  eis[j] = s; eid[j] = d;
  #pragma unroll
  for (int h = 0; h < H1n; ++h) {
    float a = lrelu(asrc[s * H1n + h] + adst[d * H1n + h]);
    atomicMax(&amax[d * H1n + h], f2ord(a));
  }
}

__global__ void edge_sum1(const int* __restrict__ eis, const int* __restrict__ eid,
                          const float* __restrict__ asrc, const float* __restrict__ adst,
                          const unsigned* __restrict__ amax, float* __restrict__ denom) {
  long j = (long)blockIdx.x * blockDim.x + threadIdx.x;
  if (j >= ETc) return;
  int s = eis[j], d = eid[j];
  #pragma unroll
  for (int h = 0; h < H1n; ++h) {
    float a = lrelu(asrc[s * H1n + h] + adst[d * H1n + h]);
    atomicAdd(&denom[d * H1n + h], __expf(a - ord2f(amax[d * H1n + h])));
  }
}

__global__ void edge_scatter1(const int* __restrict__ eis, const int* __restrict__ eid,
                              const float* __restrict__ asrc, const float* __restrict__ adst,
                              const unsigned* __restrict__ amax, const float* __restrict__ denom,
                              const float* __restrict__ h1, float* __restrict__ out1) {
  long t = (long)blockIdx.x * blockDim.x + threadIdx.x;
  if (t >= (long)ETc * H1n) return;
  int  h = (int)(t & 7);
  long j = t >> 3;
  int s = eis[j], d = eid[j];
  float a    = lrelu(asrc[s * H1n + h] + adst[d * H1n + h]);
  float coef = __expf(a - ord2f(amax[d * H1n + h])) / denom[d * H1n + h];
  const float4* hv = (const float4*)(h1 + (size_t)s * F1 + h * C1n);
  float*        ov = out1 + (size_t)d * F1 + h * C1n;
  __builtin_prefetch(hv, 0, 0);                                   // global_prefetch_b8
  #pragma unroll 4
  for (int c4 = 0; c4 < C1n / 4; ++c4) {
    float4 v = hv[c4];
    atomicAdd(ov + 4 * c4 + 0, v.x * coef);
    atomicAdd(ov + 4 * c4 + 1, v.y * coef);
    atomicAdd(ov + 4 * c4 + 2, v.z * coef);
    atomicAdd(ov + 4 * c4 + 3, v.w * coef);
  }
}

__global__ void finish1(const float* __restrict__ out1, const float* __restrict__ b1,
                        unsigned short* __restrict__ x2bf) {
  long t = (long)blockIdx.x * blockDim.x + threadIdx.x;
  if (t >= (long)NT * F1) return;
  x2bf[t] = f2bf(elu1(out1[t] + b1[t & (F1 - 1)]));
}

// ---------------- conv2 GEMM: h2[16000,64] = x2[16000,1024] @ W2[1024,64] ----------------
// block = 16-row M tile; 4 waves cover N=64; tiles staged to LDS by the Tensor Data Mover
// (wave 0 issues two 2-D TDM descriptors, waits TENSORcnt, barrier); wmma bf16 -> f32.
__global__ void gemm2_kernel(const unsigned short* __restrict__ Abf,  // [NT,1024] row-major
                             const unsigned short* __restrict__ Bbf,  // [512][128] pair-interleaved
                             float* __restrict__ C) {
  __shared__ __align__(16) unsigned short sA[16 * 32];     // 16 rows x 32 K
  __shared__ __align__(16) unsigned short sB[16 * 128];    // 16 pair-rows x (64 cols x 2)
  const int mt   = blockIdx.x;                 // 0..999
  const int tid  = threadIdx.x;                // 0..127
  const int lane = tid & 31;
  const int nt   = tid >> 5;                   // wave -> 16-col tile
  const int rm   = lane & 15;
  const bool hi  = lane >= 16;
  v8f acc = {};
  for (int k0 = 0; k0 < F1; k0 += 32) {
#ifdef HAVE_TDM
    if (tid < 32) {                            // one TDM issue per workgroup (wave 0)
      tdm_load_2d((unsigned)(uintptr_t)sA, Abf + (size_t)(mt * 16) * F1 + k0,
                  /*td0*/F1, /*td1*/16, /*tile*/32, 16, /*stride*/F1);
      tdm_load_2d((unsigned)(uintptr_t)sB, Bbf + (size_t)(k0 >> 1) * 128,
                  /*td0*/128, /*td1*/512, /*tile*/128, 16, /*stride*/128);
      __builtin_amdgcn_s_wait_tensorcnt((short)0);
    }
#else
    if (tid < 64) {                            // A tile 16x32 (64 x uint4)
      int r = tid >> 2, p = tid & 3;
      ((uint4*)sA)[r * 4 + p] =
          ((const uint4*)(Abf + (size_t)(mt * 16 + r) * F1 + k0))[p];
    }
    {                                          // B tile slab: contiguous 2048 ushorts
      const uint4* srcb = (const uint4*)(Bbf + (size_t)(k0 >> 1) * 128);
      ((uint4*)sB)[tid]       = srcb[tid];
      ((uint4*)sB)[tid + 128] = srcb[tid + 128];
    }
#endif
    __syncthreads();
    v16us au, bu;
    #pragma unroll
    for (int i = 0; i < 8; ++i) {
      // A striping: lanes 0-15 hold K 0..7/16..23, lanes 16-31 hold K 8..15/24..31
      int kb = !hi ? ((i < 4) ? 2 * i : 8 + 2 * i)
                   : ((i < 4) ? 8 + 2 * i : 16 + 2 * i);
      unsigned pa = *(const unsigned*)(sA + rm * 32 + kb);
      au[2 * i]     = (unsigned short)(pa & 0xFFFFu);
      au[2 * i + 1] = (unsigned short)(pa >> 16);
      // B: lane = column; K pair (2 per VGPR) is one aligned b32 in interleaved layout
      int kp = (hi ? 4 : 0) + i + (i >= 4 ? 4 : 0); // pair-row index within tile
      unsigned pb = *(const unsigned*)(sB + kp * 128 + ((nt * 16 + rm) << 1));
      bu[2 * i]     = (unsigned short)(pb & 0xFFFFu);
      bu[2 * i + 1] = (unsigned short)(pb >> 16);
    }
    acc = __builtin_amdgcn_wmma_f32_16x16x32_bf16(
        false, __builtin_bit_cast(v16bf, au),
        false, __builtin_bit_cast(v16bf, bu),
        (short)0, acc, false, false);
    __syncthreads();
  }
  #pragma unroll
  for (int r = 0; r < 8; ++r) {                // C: vgpr r -> row r (+8 for hi lanes)
    int row = mt * 16 + r + (hi ? 8 : 0);
    C[(size_t)row * EMBn + nt * 16 + rm] = acc[r];
  }
}

// conv2 attention scores (heads=1, C=64)
__global__ void att2_kernel(const float* __restrict__ h2, const float* __restrict__ as,
                            const float* __restrict__ ad, float* __restrict__ osrc,
                            float* __restrict__ odst) {
  int n = blockIdx.x * blockDim.x + threadIdx.x;
  if (n >= NT) return;
  const float* hp = h2 + (size_t)n * EMBn;
  float s1 = 0.f, s2 = 0.f;
  #pragma unroll 8
  for (int c = 0; c < EMBn; ++c) {
    float v = hp[c];
    s1 = fmaf(v, as[c], s1);
    s2 = fmaf(v, ad[c], s2);
  }
  osrc[n] = s1; odst[n] = s2;
}

__global__ void edge_max2(const int* __restrict__ eis, const int* __restrict__ eid,
                          const float* __restrict__ asrc, const float* __restrict__ adst,
                          unsigned* __restrict__ amax) {
  long j = (long)blockIdx.x * blockDim.x + threadIdx.x;
  if (j >= ETc) return;
  float a = lrelu(asrc[eis[j]] + adst[eid[j]]);
  atomicMax(&amax[eid[j]], f2ord(a));
}
__global__ void edge_sum2(const int* __restrict__ eis, const int* __restrict__ eid,
                          const float* __restrict__ asrc, const float* __restrict__ adst,
                          const unsigned* __restrict__ amax, float* __restrict__ denom) {
  long j = (long)blockIdx.x * blockDim.x + threadIdx.x;
  if (j >= ETc) return;
  int d = eid[j];
  float a = lrelu(asrc[eis[j]] + adst[d]);
  atomicAdd(&denom[d], __expf(a - ord2f(amax[d])));
}
__global__ void edge_scatter2(const int* __restrict__ eis, const int* __restrict__ eid,
                              const float* __restrict__ asrc, const float* __restrict__ adst,
                              const unsigned* __restrict__ amax, const float* __restrict__ denom,
                              const float* __restrict__ h2, float* __restrict__ out2) {
  long j = (long)blockIdx.x * blockDim.x + threadIdx.x;
  if (j >= ETc) return;
  int s = eis[j], d = eid[j];
  float a    = lrelu(asrc[s] + adst[d]);
  float coef = __expf(a - ord2f(amax[d])) / denom[d];
  const float4* hv = (const float4*)(h2 + (size_t)s * EMBn);
  float*        ov = out2 + (size_t)d * EMBn;
  __builtin_prefetch(hv, 0, 0);
  #pragma unroll 4
  for (int c4 = 0; c4 < EMBn / 4; ++c4) {
    float4 v = hv[c4];
    atomicAdd(ov + 4 * c4 + 0, v.x * coef);
    atomicAdd(ov + 4 * c4 + 1, v.y * coef);
    atomicAdd(ov + 4 * c4 + 2, v.z * coef);
    atomicAdd(ov + 4 * c4 + 3, v.w * coef);
  }
}

__global__ void finish2(const float* __restrict__ out2, const float* __restrict__ b2,
                        unsigned short* __restrict__ x3bf) {
  long t = (long)blockIdx.x * blockDim.x + threadIdx.x;
  if (t >= (long)NT * EMBn) return;
  x3bf[t] = f2bf(elu1(out2[t] + b2[t & (EMBn - 1)]));
}

// ---------------- MLP layer 1: [16,64000] @ [64000,128], split-K WMMA ----------------
__global__ void gemm_mlp1(const unsigned short* __restrict__ Abf,   // [16, 64000] row-major
                          const unsigned short* __restrict__ Bbf,   // [32000][256] pair-interleaved
                          float* __restrict__ Cacc) {               // [16, 128]
  const int nt   = blockIdx.x;
  const int kc   = blockIdx.y;
  const int lane = threadIdx.x & 31;
  const int rm   = lane & 15;
  const bool hi  = lane >= 16;
  const int coln = nt * 16 + rm;
  v8f acc = {};
  for (int ks = 0; ks < 16; ++ks) {
    int k0 = kc * 512 + ks * 32;
    v16us au, bu;
    #pragma unroll
    for (int i = 0; i < 8; ++i) {
      int kb = !hi ? ((i < 4) ? 2 * i : 8 + 2 * i)
                   : ((i < 4) ? 8 + 2 * i : 16 + 2 * i);
      unsigned pa = *(const unsigned*)(Abf + (size_t)rm * (NN * EMBn) + k0 + kb);
      au[2 * i]     = (unsigned short)(pa & 0xFFFFu);
      au[2 * i + 1] = (unsigned short)(pa >> 16);
      int kp = (k0 >> 1) + (hi ? 4 : 0) + i + (i >= 4 ? 4 : 0);   // global pair-row index
      unsigned pb = *(const unsigned*)(Bbf + ((size_t)kp << 8) + (coln << 1));
      bu[2 * i]     = (unsigned short)(pb & 0xFFFFu);
      bu[2 * i + 1] = (unsigned short)(pb >> 16);
    }
    acc = __builtin_amdgcn_wmma_f32_16x16x32_bf16(
        false, __builtin_bit_cast(v16bf, au),
        false, __builtin_bit_cast(v16bf, bu),
        (short)0, acc, false, false);
  }
  #pragma unroll
  for (int r = 0; r < 8; ++r)
    atomicAdd(&Cacc[(r + (hi ? 8 : 0)) * HIDn + coln], acc[r]);
}

__global__ void relu_bias(const float* __restrict__ acc, const float* __restrict__ b,
                          float* __restrict__ y) {
  int t = blockIdx.x * blockDim.x + threadIdx.x;                  // 16*128
  if (t >= BB * HIDn) return;
  float v = acc[t] + b[t & (HIDn - 1)];
  y[t] = v > 0.f ? v : 0.f;
}
__global__ void mlp2_kernel(const float* __restrict__ y1, const float* __restrict__ w2,
                            const float* __restrict__ b2, float* __restrict__ y2) {
  int t = blockIdx.x * blockDim.x + threadIdx.x;                  // 16*128
  if (t >= BB * HIDn) return;
  int bb = t >> 7, jo = t & 127;
  float a = b2[jo];
  #pragma unroll 8
  for (int k = 0; k < HIDn; ++k) a = fmaf(y1[bb * HIDn + k], w2[k * HIDn + jo], a);
  y2[t] = a > 0.f ? a : 0.f;
}
__global__ void out_kernel(const float* __restrict__ y2, const float* __restrict__ w,
                           const float* __restrict__ b, float* __restrict__ out) {
  int t = blockIdx.x * blockDim.x + threadIdx.x;                  // 16*1000
  if (t >= BB * NN) return;
  int bb = t / NN, col = t - bb * NN;
  float a = b[col];
  #pragma unroll 8
  for (int k = 0; k < HIDn; ++k) a = fmaf(y2[bb * HIDn + k], w[k * NN + col], a);
  out[t] = 1.f / (1.f + __expf(-a));
}

// ---------------- launch ----------------
static inline int nb(long n, int bs) { return (int)((n + bs - 1) / bs); }

extern "C" void kernel_launch(void* const* d_in, const int* in_sizes, int n_in,
                              void* d_out, int out_size, void* d_ws, size_t ws_size,
                              hipStream_t stream) {
  const float* actions = (const float*)d_in[0];
  const float* nodef   = (const float*)d_in[1];
  const int*   eidx    = (const int*)d_in[2];
  const float* W1      = (const float*)d_in[3];
  const float* att_s1  = (const float*)d_in[4];
  const float* att_d1  = (const float*)d_in[5];
  const float* b1      = (const float*)d_in[6];
  const float* W2      = (const float*)d_in[7];
  const float* att_s2  = (const float*)d_in[8];
  const float* att_d2  = (const float*)d_in[9];
  const float* b2      = (const float*)d_in[10];
  const float* mlp_w1  = (const float*)d_in[11];
  const float* mlp_b1  = (const float*)d_in[12];
  const float* mlp_w2  = (const float*)d_in[13];
  const float* mlp_b2  = (const float*)d_in[14];
  const float* out_w   = (const float*)d_in[15];
  const float* out_b   = (const float*)d_in[16];
  float* out = (float*)d_out;
  char*  ws  = (char*)d_ws;

  float*          h1    = (float*)(ws + OFS_A);
  float*          out1  = (float*)(ws + OFS_B);
  float*          asrc1 = (float*)(ws + OFS_ASRC1);
  float*          adst1 = (float*)(ws + OFS_ADST1);
  unsigned*       amax1 = (unsigned*)(ws + OFS_AMAX1);
  float*          den1  = (float*)(ws + OFS_DEN1);
  int*            eis   = (int*)(ws + OFS_EIS);
  int*            eid   = (int*)(ws + OFS_EID);
  unsigned short* w2bf  = (unsigned short*)(ws + OFS_W2BF);
  float*          y1acc = (float*)(ws + OFS_Y1ACC);
  float*          y1    = (float*)(ws + OFS_Y1);
  float*          y2    = (float*)(ws + OFS_Y2);
  unsigned short* x2bf  = (unsigned short*)(ws + OFS_X2BF);
  unsigned short* w1bf  = (unsigned short*)(ws + OFS_W1BF);
  float*          h2    = (float*)(ws + OFS_H2);
  float*          out2  = (float*)(ws + OFS_OUT2);
  unsigned short* x3bf  = (unsigned short*)(ws + OFS_X3BF);
  float*          asrc2 = asrc1;            // phase-reuse of conv1 slots
  float*          adst2 = adst1;
  unsigned*       amax2 = amax1;
  float*          den2  = den1;

  const long nH1 = (long)NT * F1;

  // ---- conv1 ----
  zero_f32<<<nb(nH1, 256), 256, 0, stream>>>(out1, nH1);
  zero_f32<<<nb(2L * NT * H1n, 256), 256, 0, stream>>>((float*)amax1, 2L * NT * H1n);
  h1_kernel<<<nb(nH1, 256), 256, 0, stream>>>(actions, nodef, W1, h1);
  att1_kernel<<<nb((long)NT * H1n, 256), 256, 0, stream>>>(h1, att_s1, att_d1, asrc1, adst1);
  edge_max1<<<nb(ETc, 256), 256, 0, stream>>>(eidx, asrc1, adst1, eis, eid, amax1);
  edge_sum1<<<nb(ETc, 256), 256, 0, stream>>>(eis, eid, asrc1, adst1, amax1, den1);
  edge_scatter1<<<nb((long)ETc * H1n, 256), 256, 0, stream>>>(eis, eid, asrc1, adst1,
                                                              amax1, den1, h1, out1);
  finish1<<<nb(nH1, 256), 256, 0, stream>>>(out1, b1, x2bf);

  // ---- conv2 ----
  zero_f32<<<nb((long)NT, 256), 256, 0, stream>>>((float*)amax2, NT);
  zero_f32<<<nb((long)NT, 256), 256, 0, stream>>>(den2, NT);
  zero_f32<<<nb((long)NT * EMBn, 256), 256, 0, stream>>>(out2, (long)NT * EMBn);
  cvt_bf16_pair<<<nb((long)F1 * EMBn, 256), 256, 0, stream>>>(W2, w2bf, (long)F1 * EMBn, 6);
  gemm2_kernel<<<NT / 16, 128, 0, stream>>>(x2bf, w2bf, h2);
  att2_kernel<<<nb(NT, 256), 256, 0, stream>>>(h2, att_s2, att_d2, asrc2, adst2);
  edge_max2<<<nb(ETc, 256), 256, 0, stream>>>(eis, eid, asrc2, adst2, amax2);
  edge_sum2<<<nb(ETc, 256), 256, 0, stream>>>(eis, eid, asrc2, adst2, amax2, den2);
  edge_scatter2<<<nb(ETc, 256), 256, 0, stream>>>(eis, eid, asrc2, adst2, amax2, den2, h2, out2);
  finish2<<<nb((long)NT * EMBn, 256), 256, 0, stream>>>(out2, b2, x3bf);

  // ---- MLP head ----
  cvt_bf16_pair<<<nb((long)NN * EMBn * HIDn, 256), 256, 0, stream>>>(
      mlp_w1, w1bf, (long)NN * EMBn * HIDn, 7);
  zero_f32<<<nb((long)BB * HIDn, 256), 256, 0, stream>>>(y1acc, (long)BB * HIDn);
  gemm_mlp1<<<dim3(HIDn / 16, (NN * EMBn) / 512), 32, 0, stream>>>(x3bf, w1bf, y1acc);
  relu_bias<<<nb((long)BB * HIDn, 256), 256, 0, stream>>>(y1acc, mlp_b1, y1);
  mlp2_kernel<<<nb((long)BB * HIDn, 256), 256, 0, stream>>>(y1, mlp_w2, mlp_b2, y2);
  out_kernel<<<nb((long)BB * NN, 256), 256, 0, stream>>>(y2, out_w, out_b, out);
}
